// MLPProjectionFilter_59038620451427
// MI455X (gfx1250) — matrix operand order repaired and checked by reading, not compiled, not run on hardware
//
#include <hip/hip_runtime.h>
#include <math.h>

// ---------------- problem constants ----------------
#define BATCH    1024
#define NVAR     700
#define HIDDEN   1024
#define NSTEPS   100
#define NDOF     7
#define NVC      1400
#define NAC      1386
#define NJC      1372
#define MCON     5558          // NVC+NAC+NJC+NPC
#define OUTDIM   6958          // 2*NVAR + MCON
#define NKKT     707           // NVAR + NDOF
#define NAUG     1414          // 2*NKKT
#define TSTEP    0.02f
#define INVT     50.0f
#define MAXIT    20

// d_out layout (floats)
#define OFF_XI      0            // 1024*700
#define OFF_AVGF    716800       // 1024
#define OFF_AVGP    717824       // 1024
#define OFF_PRIMAL  718848       // 20*1024
#define OFF_FIXED   739328       // 20*1024

typedef __attribute__((ext_vector_type(16))) __bf16 bfvec16;
typedef __attribute__((ext_vector_type(8)))  float  fvec8;

__device__ __forceinline__ fvec8 wmma_bf16(bfvec16 a, bfvec16 b, fvec8 c) {
  return __builtin_amdgcn_wmma_f32_16x16x32_bf16(false, a, false, b, (short)0, c, false, false);
}

// f32 -> bf16 hi/lo split (bit patterns), for split-3 fp32-fidelity WMMA
__device__ __forceinline__ void bf16_split(float x, unsigned short& hb, unsigned short& lb) {
  __bf16 h = (__bf16)x;
  __bf16 l = (__bf16)(x - (float)h);
  hb = __builtin_bit_cast(unsigned short, h);
  lb = __builtin_bit_cast(unsigned short, l);
}

// closed-form b_control(c, theta)
__device__ __forceinline__ float bctl(int c, float theta) {
  if (c < NVC)                 return 1.0f;        // V_MAX
  if (c < NVC + NAC)           return 2.0f;        // A_MAX
  if (c < NVC + NAC + NJC)     return 5.0f;        // J_MAX
  if (c < NVC + NAC + NJC + 700) return 3.14f - theta;
  return 3.14f + theta;
}

// ---------------- A_control builder (closed form) ----------------
__global__ __launch_bounds__(256) void build_actl(float* __restrict__ A, float* __restrict__ AT) {
  int idx = blockIdx.x * 256 + threadIdx.x;
  if (idx >= MCON * NVAR) return;
  int r = idx / NVAR, c = idx % NVAR;
  int d = c / NSTEPS, t = c % NSTEPS;
  float val = 0.0f;
  if (r < NVC) {                                   // velocity: +/- I
    int rr = r;             int dd = rr / 200; int q = rr % 200;
    float sgn = (q < 100) ? 1.0f : -1.0f;          int i = q % 100;
    if (dd == d && t == i) val = sgn;
  } else if (r < NVC + NAC) {                      // acceleration: diff/T
    int rr = r - NVC;       int dd = rr / 198; int q = rr % 198;
    float sgn = (q < 99) ? 1.0f : -1.0f;           int i = q % 99;
    if (dd == d) { if (t == i) val = -sgn * INVT; else if (t == i + 1) val = sgn * INVT; }
  } else if (r < NVC + NAC + NJC) {                // jerk: diff^2/T^2
    int rr = r - NVC - NAC; int dd = rr / 196; int q = rr % 196;
    float sgn = (q < 98) ? 1.0f : -1.0f;           int i = q % 98;
    float c2 = INVT * INVT;
    if (dd == d) {
      if (t == i) val = sgn * c2;
      else if (t == i + 1) val = -2.0f * sgn * c2;
      else if (t == i + 2) val = sgn * c2;
    }
  } else {                                         // position: cumsum*T
    int rr = r - NVC - NAC - NJC; int dd = rr / 200; int q = rr % 200;
    float sgn = (q < 100) ? 1.0f : -1.0f;          int i = q % 100;
    if (dd == d && t <= i) val = sgn * TSTEP;
  }
  A [(size_t)r * NVAR + c] = val;
  AT[(size_t)c * MCON + r] = val;
}

// ---------------- KKT assembly + Gauss-Jordan ----------------
__global__ __launch_bounds__(256) void assemble_q(float* __restrict__ Qaug, const float* __restrict__ AtA) {
  int idx = blockIdx.x * 256 + threadIdx.x;
  if (idx >= NKKT * NAUG) return;
  int i = idx / NAUG, j = idx % NAUG;
  float v = 0.0f;
  if (j < NKKT) {
    if (i < NVAR && j < NVAR) { v = AtA[(size_t)i * NVAR + j]; if (i == j) v += 1.0f; } // I + rho*AtA
    else if (i < NVAR)        { v = (i == (j - NVAR) * NSTEPS) ? 1.0f : 0.0f; }          // A_eq^T
    else if (j < NVAR)        { v = (j == (i - NVAR) * NSTEPS) ? 1.0f : 0.0f; }          // A_eq
  } else {
    v = ((j - NKKT) == i) ? 1.0f : 0.0f;                                                 // identity (augmented)
  }
  Qaug[idx] = v;
}

__global__ __launch_bounds__(256) void gj_elim(float* __restrict__ M, int p) {
  int i = blockIdx.x;                 // one block owns one row: no cross-block races
  if (i == p) return;
  __shared__ float f;
  if (threadIdx.x == 0) f = M[(size_t)i * NAUG + p] / M[(size_t)p * NAUG + p];
  __syncthreads();
  float ff = f;
  for (int c = threadIdx.x; c < NAUG; c += 256)
    M[(size_t)i * NAUG + c] -= ff * M[(size_t)p * NAUG + c];
}

__global__ __launch_bounds__(256) void gj_finish(const float* __restrict__ Maug, float* __restrict__ Qinv) {
  int idx = blockIdx.x * 256 + threadIdx.x;
  if (idx >= NKKT * NKKT) return;
  int i = idx / NKKT, j = idx % NKKT;
  Qinv[idx] = Maug[(size_t)i * NAUG + NKKT + j] / Maug[(size_t)i * NAUG + i];
}

// ---------------- quantile normalization (bitonic sort per column) ----------------
__global__ __launch_bounds__(1024) void quantile_norm(const float* __restrict__ in, float* __restrict__ outp) {
  int col = blockIdx.x;
  int tid = threadIdx.x;
  __shared__ float sh[1024];
  __shared__ float stats[2];
  float v = in[(size_t)tid * NVAR + col];
  sh[tid] = v;
  __syncthreads();
  for (int k = 2; k <= 1024; k <<= 1) {
    for (int j = k >> 1; j > 0; j >>= 1) {
      int ixj = tid ^ j;
      if (ixj > tid) {
        float a = sh[tid], b = sh[ixj];
        bool up = ((tid & k) == 0);
        if ((a > b) == up) { sh[tid] = b; sh[ixj] = a; }
      }
      __syncthreads();
    }
  }
  if (tid == 0) {
    float med = sh[511];                               // method='lower', q=0.5, n=1024
    float q1  = sh[255] + 0.75f * (sh[256] - sh[255]); // linear, pos=255.75
    float q3  = sh[767] + 0.25f * (sh[768] - sh[767]); // linear, pos=767.25
    float iqr = q3 - q1;
    if (iqr == 0.0f) iqr = 1.0f;
    stats[0] = med; stats[1] = iqr;
  }
  __syncthreads();
  outp[(size_t)tid * NVAR + col] = (v - stats[0]) / stats[1];
}

// ---------------- generic WMMA GEMM, bf16 split-3, f32 accumulate ----------------
// Block tile 128x64, K-step 32. 8 waves; each wave owns a 32x32 output patch
// (4 accumulators -> 12 v_wmma per K-chunk). f32->bf16 hi/lo conversion happens
// ONCE at LDS staging (packed 2xbf16 per b32 store); the MAC loop is pure packed
// ds_load_b32 fragment gathers + WMMA. B is staged transposed so each fragment's
// K-run is contiguous.
// mode 0: C = A*B
// mode 1: C = relu(A*B + bias[n])
// mode 2: C = A*B + bias[n]
// mode 3: C = src - A*B          (lam update, RHO=1)
#define LDK 36  // padded row stride (shorts): pair loads land on distinct banks
__global__ __launch_bounds__(256)
void gemm_wmma(const float* __restrict__ A, int lda,
               const float* __restrict__ B, int ldb,
               float* __restrict__ C, int ldc,
               int M, int N, int K, int mode,
               const float* __restrict__ bias,
               const float* __restrict__ src, int ldsrc)
{
  __shared__ __align__(16) unsigned short AsH[128][LDK];  // bf16 bits, A hi
  __shared__ __align__(16) unsigned short AsL[128][LDK];  // bf16 bits, A lo
  __shared__ __align__(16) unsigned short BsH[64][LDK];   // bf16 bits, B^T hi
  __shared__ __align__(16) unsigned short BsL[64][LDK];   // bf16 bits, B^T lo

  const int tid  = threadIdx.x;
  const int lane = tid & 31;
  const int wave = tid >> 5;
  const int row0 = blockIdx.y * 128;
  const int col0 = blockIdx.x * 64;
  const int m0   = (wave & 3) * 32;   // 4 wave-rows of 32
  const int n0   = (wave >> 2) * 32;  // 2 wave-cols of 32
  const int half = lane >> 4;
  const int lm   = lane & 15;

  fvec8 acc[2][2] = {{{}, {}}, {{}, {}}};
  union Frag { bfvec16 v; unsigned int w[8]; };

  for (int kk = 0; kk < K; kk += 32) {
    // ---- stage A tile (128 x 32) as bf16 hi/lo, packed pair stores ----
    for (int i = tid; i < 128 * 16; i += 256) {
      int r = i >> 4, j = i & 15;          // pair j covers k = 2j, 2j+1
      int gr = row0 + r, gk = kk + 2 * j;
      float x0 = (gr < M && gk     < K) ? A[(size_t)gr * lda + gk]     : 0.0f;
      float x1 = (gr < M && gk + 1 < K) ? A[(size_t)gr * lda + gk + 1] : 0.0f;
      unsigned short h0, l0, h1, l1;
      bf16_split(x0, h0, l0); bf16_split(x1, h1, l1);
      *(unsigned int*)&AsH[r][2 * j] = (unsigned int)h0 | ((unsigned int)h1 << 16);
      *(unsigned int*)&AsL[r][2 * j] = (unsigned int)l0 | ((unsigned int)l1 << 16);
    }
    // ---- stage B tile (32 x 64) transposed as bf16 hi/lo ----
    for (int i = tid; i < 16 * 64; i += 256) {
      int kp = i >> 6, n = i & 63;         // pair kp covers k = 2kp, 2kp+1
      int gk = kk + 2 * kp, gn = col0 + n;
      float y0 = (gk     < K && gn < N) ? B[(size_t)(gk)     * ldb + gn] : 0.0f;
      float y1 = (gk + 1 < K && gn < N) ? B[(size_t)(gk + 1) * ldb + gn] : 0.0f;
      unsigned short h0, l0, h1, l1;
      bf16_split(y0, h0, l0); bf16_split(y1, h1, l1);
      *(unsigned int*)&BsH[n][2 * kp] = (unsigned int)h0 | ((unsigned int)h1 << 16);
      *(unsigned int*)&BsL[n][2 * kp] = (unsigned int)l0 | ((unsigned int)l1 << 16);
    }
    // prefetch next K-chunk while this one computes
    if (kk + 32 < K) {
      int pr = tid >> 1, pk = kk + 32 + (tid & 1) * 16;
      if (row0 + pr < M && pk < K) __builtin_prefetch(&A[(size_t)(row0 + pr) * lda + pk], 0, 3);
      int bk = kk + 32 + (tid >> 3), bn = col0 + (tid & 7) * 8;
      if (bk < K && bn < N) __builtin_prefetch(&B[(size_t)bk * ldb + bn], 0, 3);
    }
    __syncthreads();

    // ---- gather fragments: pure packed b32 LDS loads (ISA 7.12.2 layouts) ----
    Frag aH[2], aL[2], bH[2], bL[2];
#pragma unroll
    for (int u = 0; u < 8; ++u) {
      int ka = (u < 4) ? (2 * u + 8 * half) : (16 + 2 * (u - 4) + 8 * half);
#pragma unroll
      for (int mi = 0; mi < 2; ++mi) {
        int m = m0 + mi * 16 + lm;
        aH[mi].w[u] = *(const unsigned int*)&AsH[m][ka];
        aL[mi].w[u] = *(const unsigned int*)&AsL[m][ka];
      }
      int kb = (u < 4) ? (2 * u + 8 * half) : (16 + 2 * (u - 4) + 8 * half);
#pragma unroll
      for (int ni = 0; ni < 2; ++ni) {
        int n = n0 + ni * 16 + lm;
        bH[ni].w[u] = *(const unsigned int*)&BsH[n][kb];
        bL[ni].w[u] = *(const unsigned int*)&BsL[n][kb];
      }
    }
    // split-3 per tile: hi*hi + lo*hi + hi*lo (~fp32 fidelity)
#pragma unroll
    for (int mi = 0; mi < 2; ++mi)
#pragma unroll
      for (int ni = 0; ni < 2; ++ni) {
        acc[mi][ni] = wmma_bf16(aH[mi].v, bH[ni].v, acc[mi][ni]);
        acc[mi][ni] = wmma_bf16(aL[mi].v, bH[ni].v, acc[mi][ni]);
        acc[mi][ni] = wmma_bf16(aH[mi].v, bL[ni].v, acc[mi][ni]);
      }
    __syncthreads();
  }

  // ---- epilogue ----
#pragma unroll
  for (int mi = 0; mi < 2; ++mi) {
#pragma unroll
    for (int r = 0; r < 8; ++r) {
      int row = row0 + m0 + mi * 16 + r + 8 * half;  // D layout: VGPR r -> M=r / M=r+8
      if (row >= M) continue;
#pragma unroll
      for (int ni = 0; ni < 2; ++ni) {
        int col = col0 + n0 + ni * 16 + lm;
        if (col >= N) continue;
        float o = acc[mi][ni][r];
        if (mode == 1)      { o += bias[col]; o = fmaxf(o, 0.0f); }
        else if (mode == 2) { o += bias[col]; }
        else if (mode == 3) { o = src[(size_t)row * ldsrc + col] - o; }
        C[(size_t)row * ldc + col] = o;
      }
    }
  }
}

// ---------------- elementwise / epilogue kernels ----------------
__global__ __launch_bounds__(256) void split_out(const float* __restrict__ o,
                                                 float* __restrict__ lam, float* __restrict__ s) {
  int idx = blockIdx.x * 256 + threadIdx.x;
  if (idx >= BATCH * (NVAR + MCON)) return;
  int b = idx / (NVAR + MCON), j = idx % (NVAR + MCON);
  if (j < NVAR) lam[(size_t)b * NVAR + j] = o[(size_t)b * OUTDIM + NVAR + j];
  else {
    int c = j - NVAR;
    s[(size_t)b * MCON + c] = fmaxf(0.0f, o[(size_t)b * OUTDIM + 2 * NVAR + c]);
  }
}

__global__ __launch_bounds__(256) void baug_kernel(float* __restrict__ baug,
                                                   const float* __restrict__ s,
                                                   const float* __restrict__ theta) {
  int idx = blockIdx.x * 256 + threadIdx.x;
  if (idx >= BATCH * MCON) return;
  int b = idx / MCON, c = idx % MCON;
  baug[idx] = bctl(c, theta[b]) - s[idx];
}

__global__ __launch_bounds__(256) void rhs_kernel(float* __restrict__ rhs,
                                                  const float* __restrict__ lam,
                                                  const float* __restrict__ inp,
                                                  const float* __restrict__ t700,
                                                  const float* __restrict__ vst) {
  int idx = blockIdx.x * 256 + threadIdx.x;
  if (idx >= BATCH * NKKT) return;
  int b = idx / NKKT, j = idx % NKKT;
  // rhs = -lincost = lam + xi_samples + rho*(b_aug@A), tail = b_eq
  rhs[idx] = (j < NVAR)
           ? (lam[(size_t)b * NVAR + j] + inp[(size_t)b * NVAR + j] + t700[(size_t)b * NVAR + j])
           : vst[(size_t)b * NDOF + (j - NVAR)];
}

// Ax -> (s_new, res) in place: s_new = max(0, b - Ax); res = Ax - b + s_new
__global__ __launch_bounds__(256) void snew_kernel(float* __restrict__ axres,
                                                   float* __restrict__ s_new,
                                                   const float* __restrict__ theta) {
  int idx = blockIdx.x * 256 + threadIdx.x;
  if (idx >= BATCH * MCON) return;
  int b = idx / MCON, c = idx % MCON;
  float bc = bctl(c, theta[b]);
  float ax = axres[idx];
  float sn = fmaxf(0.0f, bc - ax);
  s_new[idx] = sn;
  axres[idx] = ax - bc + sn;
}

__global__ __launch_bounds__(256) void norms_kernel(const float* __restrict__ res,
                                                    const float* __restrict__ lam,
                                                    const float* __restrict__ lam_new,
                                                    const float* __restrict__ s,
                                                    const float* __restrict__ s_new,
                                                    float* __restrict__ out_primal,
                                                    float* __restrict__ out_fixed) {
  int b = blockIdx.x, tid = threadIdx.x;
  float sres = 0.0f, sl = 0.0f, ss = 0.0f;
  const float* rr = res   + (size_t)b * MCON;
  const float* sr = s     + (size_t)b * MCON;
  const float* nr = s_new + (size_t)b * MCON;
  for (int c = tid; c < MCON; c += 256) {
    float r = rr[c]; sres += r * r;
    float d = sr[c] - nr[c]; ss += d * d;
  }
  const float* lr = lam     + (size_t)b * NVAR;
  const float* ln = lam_new + (size_t)b * NVAR;
  for (int c = tid; c < NVAR; c += 256) { float d = lr[c] - ln[c]; sl += d * d; }
  __shared__ float r0[256], r1[256], r2[256];
  r0[tid] = sres; r1[tid] = sl; r2[tid] = ss;
  __syncthreads();
  for (int off = 128; off > 0; off >>= 1) {
    if (tid < off) { r0[tid] += r0[tid + off]; r1[tid] += r1[tid + off]; r2[tid] += r2[tid + off]; }
    __syncthreads();
  }
  if (tid == 0) {
    out_primal[b] = sqrtf(r0[0]);
    out_fixed[b]  = sqrtf(r1[0]) + sqrtf(r2[0]);
  }
}

__global__ __launch_bounds__(256) void copy_xi(const float* __restrict__ sol, float* __restrict__ o) {
  int idx = blockIdx.x * 256 + threadIdx.x;
  if (idx >= BATCH * NVAR) return;
  int b = idx / NVAR, j = idx % NVAR;
  o[idx] = sol[(size_t)b * NKKT + j];
}

__global__ __launch_bounds__(256) void avg_kernel(float* __restrict__ o) {
  int b = blockIdx.x * 256 + threadIdx.x;
  if (b >= BATCH) return;
  float sp = 0.0f, sf = 0.0f;
  for (int t = 0; t < MAXIT; ++t) {
    sp += o[OFF_PRIMAL + t * BATCH + b];
    sf += o[OFF_FIXED  + t * BATCH + b];
  }
  o[OFF_AVGF + b] = sf / (float)MAXIT;
  o[OFF_AVGP + b] = sp / (float)MAXIT;
}

// ---------------- host orchestration ----------------
static void launch_gemm(const float* A, int lda, const float* B, int ldb,
                        float* C, int ldc, int M, int N, int K, int mode,
                        const float* bias, const float* src, int ldsrc, hipStream_t stream) {
  dim3 grid((N + 63) / 64, (M + 127) / 128);
  gemm_wmma<<<grid, 256, 0, stream>>>(A, lda, B, ldb, C, ldc, M, N, K, mode, bias, src, ldsrc);
}

extern "C" void kernel_launch(void* const* d_in, const int* in_sizes, int n_in,
                              void* d_out, int out_size, void* d_ws, size_t ws_size,
                              hipStream_t stream) {
  (void)in_sizes; (void)n_in; (void)out_size; (void)ws_size;
  const float* input_nn = (const float*)d_in[0];
  const float* theta    = (const float*)d_in[1];
  const float* v_start  = (const float*)d_in[2];
  const float* W1       = (const float*)d_in[3];
  const float* b1       = (const float*)d_in[4];
  const float* W2       = (const float*)d_in[5];
  const float* b2       = (const float*)d_in[6];
  float* out = (float*)d_out;

  float* ws = (float*)d_ws;
  size_t off = 0;
  auto alloc = [&](size_t n) { float* p = ws + off; off += n; return p; };
  float* ACTL = alloc((size_t)MCON * NVAR);
  float* AT   = alloc((size_t)NVAR * MCON);
  float* QAUG = alloc((size_t)NKKT * NAUG);
  float* QINV = alloc((size_t)NKKT * NKKT);   // staged as AtA first
  float* INP  = alloc((size_t)BATCH * NVAR);  // inp_norm; reused as t700
  float* H    = alloc((size_t)BATCH * HIDDEN);// h; reused as rhs (BATCH*707 fits)
  float* OUTB = alloc((size_t)BATCH * OUTDIM);// MLP out; reused as b_aug
  float* LAMA = alloc((size_t)BATCH * NVAR);
  float* LAMB = alloc((size_t)BATCH * NVAR);
  float* SA   = alloc((size_t)BATCH * MCON);
  float* SB   = alloc((size_t)BATCH * MCON);
  float* SOL  = alloc((size_t)BATCH * NKKT);
  float* AXR  = alloc((size_t)BATCH * MCON);

  // ---- constants: A_control, KKT inverse ----
  {
    int tot = MCON * NVAR;
    build_actl<<<(tot + 255) / 256, 256, 0, stream>>>(ACTL, AT);
  }
  // AtA = A^T A  (700x700, K=5558) into QINV staging
  launch_gemm(AT, MCON, ACTL, NVAR, QINV, NVAR, NVAR, NVAR, MCON, 0, nullptr, nullptr, 0, stream);
  {
    int tot = NKKT * NAUG;
    assemble_q<<<(tot + 255) / 256, 256, 0, stream>>>(QAUG, QINV);
  }
  for (int p = 0; p < NKKT; ++p)
    gj_elim<<<NKKT, 256, 0, stream>>>(QAUG, p);
  gj_finish<<<(NKKT * NKKT + 255) / 256, 256, 0, stream>>>(QAUG, QINV);  // Q symmetric -> Qinv^T == Qinv

  // ---- MLP ----
  quantile_norm<<<NVAR, 1024, 0, stream>>>(input_nn, INP);
  launch_gemm(INP, NVAR, W1, HIDDEN, H, HIDDEN, BATCH, HIDDEN, NVAR, 1, b1, nullptr, 0, stream);
  launch_gemm(H, HIDDEN, W2, OUTDIM, OUTB, OUTDIM, BATCH, OUTDIM, HIDDEN, 2, b2, nullptr, 0, stream);
  split_out<<<((BATCH * (NVAR + MCON)) + 255) / 256, 256, 0, stream>>>(OUTB, LAMA, SA);

  // ---- ADMM loop ----
  float* lam  = LAMA; float* lamN = LAMB;
  float* s    = SA;   float* sN   = SB;
  float* BAUG = OUTB; float* T700 = INP; float* RHS = H;
  const int nEl = BATCH * MCON;
  for (int t = 0; t < MAXIT; ++t) {
    baug_kernel<<<(nEl + 255) / 256, 256, 0, stream>>>(BAUG, s, theta);
    launch_gemm(BAUG, MCON, ACTL, NVAR, T700, NVAR, BATCH, NVAR, MCON, 0, nullptr, nullptr, 0, stream);
    rhs_kernel<<<((BATCH * NKKT) + 255) / 256, 256, 0, stream>>>(RHS, lam, input_nn, T700, v_start);
    launch_gemm(RHS, NKKT, QINV, NKKT, SOL, NKKT, BATCH, NKKT, NKKT, 0, nullptr, nullptr, 0, stream);
    launch_gemm(SOL, NKKT, AT, MCON, AXR, MCON, BATCH, MCON, NVAR, 0, nullptr, nullptr, 0, stream);
    snew_kernel<<<(nEl + 255) / 256, 256, 0, stream>>>(AXR, sN, theta);
    launch_gemm(AXR, MCON, ACTL, NVAR, lamN, NVAR, BATCH, NVAR, MCON, 3, nullptr, lam, NVAR, stream);
    norms_kernel<<<BATCH, 256, 0, stream>>>(AXR, lam, lamN, s, sN,
                                            out + OFF_PRIMAL + t * BATCH,
                                            out + OFF_FIXED  + t * BATCH);
    { float* tmp = lam; lam = lamN; lamN = tmp; }
    { float* tmp = s;   s   = sN;   sN   = tmp; }
  }

  copy_xi<<<((BATCH * NVAR) + 255) / 256, 256, 0, stream>>>(SOL, out);
  avg_kernel<<<(BATCH + 255) / 256, 256, 0, stream>>>(out);
}
